// HierarchicalGraphIntegrator_56083682951401
// MI455X (gfx1250) — matrix operand order repaired and checked by reading, not compiled, not run on hardware
//
#include <hip/hip_runtime.h>
#include <hip/hip_bf16.h>
#include <math.h>

// ---------------------------------------------------------------------------
// Types for CDNA5 WMMA (wave32, 16x16x32 bf16 -> f32)
// ---------------------------------------------------------------------------
typedef __attribute__((ext_vector_type(16))) __bf16 v16bf;
typedef __attribute__((ext_vector_type(8)))  float  v8f;

#define ACT_NONE 0
#define ACT_RELU 1

// ---------------------------------------------------------------------------
// General strided/batched GEMM:  C[m,n] (+)= alpha * sum_k A[m,k]*B[n,k] + bias[n]
//   A[m,k] = A[m*sAm + k*sAk + z*sAb]   (row index gets +ashift, OOB -> 0)
//   B[n,k] = B[n*sBn + k*sBk + z*sBb]
//   C[m,n] = C[m*sCm + n + z*sCb]
// REQUIREMENT: K % 32 == 0 (not met by the attn @ V launch with L == 1, where K == 1).
// Block: 128x64 tile, BK=32, 256 threads = 8 waves, each wave -> 32x32 slab
// (2 A-frags x 2 B-frags -> 4 v_wmma per K-step).
// Double-buffered LDS pipeline: one barrier per K-step, global loads for the
// next tile issue before the WMMAs of the current tile.
// Out-of-range rows are handled by clamp-and-mask (no exec divergence).
// ---------------------------------------------------------------------------
__global__ __launch_bounds__(256)
void gemm_wmma_kernel(const float* __restrict__ A, long sAm, long sAk, long sAb,
                      const float* __restrict__ Bm, long sBn, long sBk, long sBb,
                      float* __restrict__ C, long sCm, long sCb,
                      const float* __restrict__ bias,
                      int M, int N, int K,
                      float alpha, int act, int accum, int ashift)
{
    // stride 40: row starts 80B apart (16B aligned), 16-row column gathers
    // land on 16 distinct banks.
    __shared__ __attribute__((aligned(16))) __bf16 lA[2][128][40];
    __shared__ __attribute__((aligned(16))) __bf16 lB[2][64][40];

    const int tid = threadIdx.x;
    const int z = blockIdx.z;
    A  += (long)z * sAb;
    Bm += (long)z * sBb;
    C  += (long)z * sCb;

    const int m0 = blockIdx.y * 128;
    const int n0 = blockIdx.x * 64;
    const int w    = tid >> 5;      // wave 0..7
    const int lane = tid & 31;
    const int l16  = lane & 15;
    const int hh   = lane >> 4;     // lane half
    const int wm = w >> 1;          // 0..3 : 32-row slab
    const int wn = w & 1;           // 0..1 : 32-col slab

    const v8f vzero = {0.f,0.f,0.f,0.f,0.f,0.f,0.f,0.f};
    v8f acc00 = vzero, acc01 = vzero, acc10 = vzero, acc11 = vzero;

    // loader indices: A 128x32 (16 elems/thread), B 64x32 (8 elems/thread)
    const int ar  = tid >> 1;          // 0..127
    const int akc = (tid & 1) * 16;    // 0 or 16
    const int br  = tid >> 2;          // 0..63
    const int bkc = (tid & 3) * 8;     // 0,8,16,24

    const bool aFast = (sAk == 1) && (m0 + ashift >= 0) && ((long)m0 + 127 + ashift < M);
    const bool bFast = (sBk == 1) && (n0 + 63 < N);

    // clamp-and-mask row bases (no exec divergence in the load path)
    int gmA = m0 + ar + ashift;
    const float aMask = (gmA >= 0 && gmA < M) ? 1.0f : 0.0f;
    int gmAc = gmA < 0 ? 0 : (gmA >= M ? (M - 1) : gmA);
    const float* aRow = A + (long)gmAc * sAm;

    int gnB = n0 + br;
    const float bMask = (gnB < N) ? 1.0f : 0.0f;
    int gnBc = gnB >= N ? (N - 1) : gnB;
    const float* bRow = Bm + (long)gnBc * sBn;

    float ta[16], tb[8];

    auto loadA = [&](int k0) {
        if (aFast) {
            const float4* ap = (const float4*)(aRow + k0 + akc);
            float4 x0 = ap[0], x1 = ap[1], x2 = ap[2], x3 = ap[3];
            ta[0]  = x0.x; ta[1]  = x0.y; ta[2]  = x0.z; ta[3]  = x0.w;
            ta[4]  = x1.x; ta[5]  = x1.y; ta[6]  = x1.z; ta[7]  = x1.w;
            ta[8]  = x2.x; ta[9]  = x2.y; ta[10] = x2.z; ta[11] = x2.w;
            ta[12] = x3.x; ta[13] = x3.y; ta[14] = x3.z; ta[15] = x3.w;
        } else {
            const float* ap = aRow + (long)(k0 + akc) * sAk;
            #pragma unroll
            for (int j = 0; j < 16; ++j) ta[j] = aMask * ap[(long)j * sAk];
        }
    };
    auto loadB = [&](int k0) {
        if (bFast) {
            const float4* bp = (const float4*)(bRow + k0 + bkc);
            float4 x0 = bp[0], x1 = bp[1];
            tb[0] = x0.x; tb[1] = x0.y; tb[2] = x0.z; tb[3] = x0.w;
            tb[4] = x1.x; tb[5] = x1.y; tb[6] = x1.z; tb[7] = x1.w;
        } else {
            const float* bp = bRow + (long)(k0 + bkc) * sBk;
            #pragma unroll
            for (int j = 0; j < 8; ++j) tb[j] = bMask * bp[(long)j * sBk];
        }
    };
    auto storeA = [&](int bi) {
        __bf16* dst = &lA[bi][ar][akc];
        #pragma unroll
        for (int j = 0; j < 16; ++j) dst[j] = (__bf16)ta[j];
    };
    auto storeB = [&](int bi) {
        __bf16* dst = &lB[bi][br][bkc];
        #pragma unroll
        for (int j = 0; j < 8; ++j) dst[j] = (__bf16)tb[j];
    };

    // prologue: stage first tile into buffer 0
    loadA(0); loadB(0);
    storeA(0); storeB(0);
    int cur = 0;

    for (int k0 = 0; k0 < K; k0 += 32) {
        __syncthreads();                     // buffer `cur` is ready
        const bool more = (k0 + 32) < K;
        if (more) { loadA(k0 + 32); loadB(k0 + 32); }   // overlap with compute

        // ---- build fragments per documented CDNA5 layouts ----
        v16bf a0, a1;
        const int m0l = wm * 32 + l16;
        const int m1l = m0l + 16;
        #pragma unroll
        for (int e = 0; e < 16; ++e) {
            int p = e >> 1, bit = e & 1;
            int kk = ((p < 4) ? (8 * hh + 2 * p) : (16 + 8 * hh + 2 * (p - 4))) + bit;
            a0[e] = lA[cur][m0l][kk];
            a1[e] = lA[cur][m1l][kk];
        }
        v16bf b0, b1;
        const int n0l = wn * 32 + l16;
        const int n1l = n0l + 16;
        #pragma unroll
        for (int e = 0; e < 16; ++e) {
            int kk = (e < 8) ? (8 * hh + e) : (16 + 8 * hh + (e - 8));
            b0[e] = lB[cur][n0l][kk];
            b1[e] = lB[cur][n1l][kk];
        }
        acc00 = __builtin_amdgcn_wmma_f32_16x16x32_bf16(false, a0, false, b0, (short)0, acc00, false, false);
        acc01 = __builtin_amdgcn_wmma_f32_16x16x32_bf16(false, a0, false, b1, (short)0, acc01, false, false);
        acc10 = __builtin_amdgcn_wmma_f32_16x16x32_bf16(false, a1, false, b0, (short)0, acc10, false, false);
        acc11 = __builtin_amdgcn_wmma_f32_16x16x32_bf16(false, a1, false, b1, (short)0, acc11, false, false);

        if (more) { storeA(cur ^ 1); storeB(cur ^ 1); }  // fill other buffer
        cur ^= 1;
    }

    // ---- epilogue: D layout row = r + 8*half, col = lane%16 ----
    #pragma unroll
    for (int ti = 0; ti < 2; ++ti) {
        #pragma unroll
        for (int tj = 0; tj < 2; ++tj) {
            v8f acc = ti ? (tj ? acc11 : acc10) : (tj ? acc01 : acc00);
            int col = n0 + wn * 32 + tj * 16 + l16;
            if (col >= N) continue;
            float bv = bias ? bias[col] : 0.0f;
            #pragma unroll
            for (int r = 0; r < 8; ++r) {
                int row = m0 + wm * 32 + ti * 16 + hh * 8 + r;
                if (row >= M) continue;
                float vv = alpha * acc[r] + bv;
                float* cp = C + (long)row * sCm + col;
                if (accum) vv += *cp;
                if (act == ACT_RELU) vv = fmaxf(vv, 0.0f);
                *cp = vv;
            }
        }
    }
}

// ---------------------------------------------------------------------------
// Elementwise / reduction kernels
// ---------------------------------------------------------------------------
__global__ void zero_kernel(float* p, long n) {
    long i = (long)blockIdx.x * 256 + threadIdx.x;
    if (i < n) p[i] = 0.0f;
}

// BatchNorm training-mode over rows, per column; x is [R, C], in-place
__global__ void bn_kernel(float* x, const float* g, const float* b, int R, int C) {
    int c = blockIdx.x;
    __shared__ float s1[256], s2[256];
    float a = 0.f, q = 0.f;
    for (int r = threadIdx.x; r < R; r += 256) {
        float v = x[(long)r * C + c];
        a += v; q += v * v;
    }
    s1[threadIdx.x] = a; s2[threadIdx.x] = q;
    __syncthreads();
    for (int s = 128; s > 0; s >>= 1) {
        if (threadIdx.x < s) { s1[threadIdx.x] += s1[threadIdx.x + s]; s2[threadIdx.x] += s2[threadIdx.x + s]; }
        __syncthreads();
    }
    float mean = s1[0] / (float)R;
    float var  = s2[0] / (float)R - mean * mean;
    float sc = g[c] * rsqrtf(var + 1e-5f);
    float sh = b[c] - mean * sc;
    for (int r = threadIdx.x; r < R; r += 256) {
        long i = (long)r * C + c;
        x[i] = x[i] * sc + sh;
    }
}

// LayerNorm over last dim (H=256), out row mapped into [B, RT, H] slices
__global__ void ln_kernel(const float* in, float* out, const float* g, const float* b,
                          int rows, int Lb, long sOb) {
    int r = blockIdx.x;
    if (r >= rows) return;
    int j = threadIdx.x;
    float v = in[(long)r * 256 + j];
    __shared__ float red[256];
    red[j] = v; __syncthreads();
    for (int s = 128; s > 0; s >>= 1) { if (j < s) red[j] += red[j + s]; __syncthreads(); }
    float mean = red[0] * (1.0f / 256.0f);
    __syncthreads();
    float d = v - mean;
    red[j] = d * d; __syncthreads();
    for (int s = 128; s > 0; s >>= 1) { if (j < s) red[j] += red[j + s]; __syncthreads(); }
    float var = red[0] * (1.0f / 256.0f);
    float o = d * rsqrtf(var + 1e-5f) * g[j] + b[j];
    out[(long)(r / Lb) * sOb + (long)(r % Lb) * 256 + j] = o;
}

// LayerNorm over the LENGTH axis of x stored [B, L, C]  (torch LN([256]) on [B,C,L])
__global__ void ln_axis_kernel(float* x, const float* g, const float* b, int L, int C) {
    int bc = blockIdx.x;
    int bb = bc / C, c = bc % C;
    int l = threadIdx.x;                 // L == 256
    long idx = (long)bb * L * C + (long)l * C + c;
    float v = x[idx];
    __shared__ float red[256];
    red[l] = v; __syncthreads();
    for (int s = 128; s > 0; s >>= 1) { if (l < s) red[l] += red[l + s]; __syncthreads(); }
    float mean = red[0] / (float)L;
    __syncthreads();
    float d = v - mean;
    red[l] = d * d; __syncthreads();
    for (int s = 128; s > 0; s >>= 1) { if (l < s) red[l] += red[l + s]; __syncthreads(); }
    float var = red[0] / (float)L;
    x[idx] = d * rsqrtf(var + 1e-5f) * g[l] + b[l];
}

__global__ void softmax_kernel(float* x, int N) {
    int r = blockIdx.x;
    float* row = x + (long)r * N;
    __shared__ float red[256];
    float m = -3.4e38f;
    for (int j = threadIdx.x; j < N; j += 256) m = fmaxf(m, row[j]);
    red[threadIdx.x] = m; __syncthreads();
    for (int s = 128; s > 0; s >>= 1) { if (threadIdx.x < s) red[threadIdx.x] = fmaxf(red[threadIdx.x], red[threadIdx.x + s]); __syncthreads(); }
    m = red[0];
    __syncthreads();
    float sum = 0.f;
    for (int j = threadIdx.x; j < N; j += 256) { float e = expf(row[j] - m); row[j] = e; sum += e; }
    red[threadIdx.x] = sum; __syncthreads();
    for (int s = 128; s > 0; s >>= 1) { if (threadIdx.x < s) red[threadIdx.x] += red[threadIdx.x + s]; __syncthreads(); }
    float inv = 1.0f / red[0];
    for (int j = threadIdx.x; j < N; j += 256) row[j] *= inv;
}

// comb[b,e,:] = concat(src[b, si[e], :], tgt[b, ti[e], :])
__global__ void gather_concat_kernel(const float* src, const float* tgt,
                                     const int* si, const int* ti,
                                     float* comb, int E, int Ls, int Lt, long total) {
    long i = (long)blockIdx.x * 256 + threadIdx.x;
    if (i >= total) return;
    int j = (int)(i & 511);
    long t = i >> 9;
    int e = (int)(t % E);
    int b = (int)(t / E);
    float v;
    if (j < 256) v = src[((long)b * Ls + si[e]) * 256 + j];
    else         v = tgt[((long)b * Lt + ti[e]) * 256 + (j - 256)];
    comb[i] = v;
}

__global__ void concat2_kernel(const float* a, const float* b, float* out, long rows) {
    long i = (long)blockIdx.x * 256 + threadIdx.x;
    long total = rows * 512;
    if (i >= total) return;
    int j = (int)(i & 511);
    long r = i >> 9;
    out[i] = (j < 256) ? a[r * 256 + j] : b[r * 256 + j - 256];
}

__global__ void scatter_add_kernel(const float* msg, const int* ti, float* dst,
                                   int E, int Lt, long total) {
    long i = (long)blockIdx.x * 256 + threadIdx.x;
    if (i >= total) return;
    int j = (int)(i & 255);
    long t = i >> 8;
    int e = (int)(t % E);
    int b = (int)(t / E);
    atomicAdd(dst + ((long)b * Lt + ti[e]) * 256 + j, msg[i]);
}

// GRUCell fuse: gi/gh are [rows,768] (biases already added in GEMM)
__global__ void gru_kernel(const float* gi, const float* gh, const float* h, float* out, long rows) {
    long i = (long)blockIdx.x * 256 + threadIdx.x;
    long total = rows * 256;
    if (i >= total) return;
    int j = (int)(i & 255);
    long r = i >> 8;
    const float* gir = gi + r * 768;
    const float* ghr = gh + r * 768;
    float rg = 1.f / (1.f + expf(-(gir[j] + ghr[j])));
    float zg = 1.f / (1.f + expf(-(gir[256 + j] + ghr[256 + j])));
    float nn = tanhf(gir[512 + j] + rg * ghr[512 + j]);
    out[i] = (1.f - zg) * nn + zg * h[i];
}

// s[i] = base[i] * sigmoid(s[i])   (in-place gate)
__global__ void gate_kernel(const float* base, float* s, long n) {
    long i = (long)blockIdx.x * 256 + threadIdx.x;
    if (i >= n) return;
    float g = 1.f / (1.f + expf(-s[i]));
    s[i] = base[i] * g;
}

// fc = concat(attn, gbu + gtd)
__global__ void fccat_kernel(const float* attn, const float* gbu, const float* gtd,
                             float* fc, long rows) {
    long i = (long)blockIdx.x * 256 + threadIdx.x;
    long total = rows * 512;
    if (i >= total) return;
    int j = (int)(i & 511);
    long r = i >> 9;
    fc[i] = (j < 256) ? attn[r * 256 + j] : (gbu[r * 256 + j - 256] + gtd[r * 256 + j - 256]);
}

// ---------------------------------------------------------------------------
// Host orchestration
// ---------------------------------------------------------------------------
static void gemm_full(hipStream_t st, const float* A, long sAm, long sAk, long sAb,
                      const float* Bm, long sBn, long sBk, long sBb,
                      float* C, long sCm, long sCb, const float* bias,
                      int M, int N, int K, int nb,
                      float alpha, int act, int accum, int ashift) {
    dim3 grid((N + 63) / 64, (M + 127) / 128, nb);
    gemm_wmma_kernel<<<grid, dim3(256), 0, st>>>(A, sAm, sAk, sAb, Bm, sBn, sBk, sBb,
                                                 C, sCm, sCb, bias, M, N, K,
                                                 alpha, act, accum, ashift);
}

static const int NPARAM = 90;
static const int PSIZES[NPARAM] = {
    8192, 128, 128, 128,
    32768, 256, 256, 256,
    98304, 256, 256, 256,
    196608, 256, 256, 256,
    65536, 256, 65536, 256, 256, 256,
    262144, 512, 131072, 256, 256, 256,
    // 6 message MLPs: W1[256,512], b1, W2[256,256], b2
    131072, 256, 65536, 256,
    131072, 256, 65536, 256,
    131072, 256, 65536, 256,
    131072, 256, 65536, 256,
    131072, 256, 65536, 256,
    131072, 256, 65536, 256,
    // 6 GRUs: Wih[768,256], Whh[768,256], bih, bhh
    196608, 196608, 768, 768,
    196608, 196608, 768, 768,
    196608, 196608, 768, 768,
    196608, 196608, 768, 768,
    196608, 196608, 768, 768,
    196608, 196608, 768, 768,
    // integrator
    196608, 768, 65536, 256,     // attn_Wi, attn_bi, attn_Wo, attn_bo
    131072, 256, 131072, 256,    // bug_W, bug_b, tdg_W, tdg_b
    131072, 256, 65536, 256,     // fin_W1, fin_b1, fin_W2, fin_b2
    256, 256                     // fin_lng, fin_lnb
};

extern "C" void kernel_launch(void* const* d_in, const int* in_sizes, int n_in,
                              void* d_out, int out_size, void* d_ws, size_t ws_size,
                              hipStream_t stream) {
    (void)in_sizes; (void)out_size; (void)ws_size;
    const int B = 4, LAA = 2048, LSS = 256, LDOM = 32, H = 256;
    const long ROWT = (long)(LAA + LSS + LDOM + 1);  // 2337 rows per batch in output

    // ---- inputs ----
    const float* aaF  = (const float*)d_in[0];
    const float* ssF  = (const float*)d_in[1];
    const float* domF = (const float*)d_in[2];
    const float* prF  = (const float*)d_in[3];
    const int* edge[12];
    for (int i = 0; i < 12; ++i) edge[i] = (const int*)d_in[4 + i];

    // ---- params: one flat buffer, or 90 separate leaves ----
    const float* P[NPARAM];
    if (n_in >= 16 + NPARAM) {
        for (int i = 0; i < NPARAM; ++i) P[i] = (const float*)d_in[16 + i];
    } else {
        const float* base = (const float*)d_in[16];
        long o = 0;
        for (int i = 0; i < NPARAM; ++i) { P[i] = base + o; o += PSIZES[i]; }
    }
    float* out = (float*)d_out;

    // ---- workspace bump allocator ----
    float* W = (float*)d_ws;
    size_t off = 0;
    auto alloc = [&](size_t n) { float* p = W + off; off += n; return p; };
    auto gb = [](long n) { return dim3((unsigned)((n + 255) / 256)); };

    float* aaN    = alloc((size_t)B * LAA * H);
    float* ssN    = alloc((size_t)B * LSS * H);
    float* domN   = alloc((size_t)B * LDOM * H);
    float* protN  = alloc((size_t)B * H);
    float* buSS   = alloc((size_t)B * LSS * H);
    float* buDOM  = alloc((size_t)B * LDOM * H);
    float* buPROT = alloc((size_t)B * H);
    float* tdDOM  = alloc((size_t)B * LDOM * H);
    float* tdSS   = alloc((size_t)B * LSS * H);
    float* tdAA   = alloc((size_t)B * LAA * H);
    const size_t scratch0 = off;

    // plain linear: C = act(A[M,K] @ Wt[N,K]^T + bias)
    auto lin = [&](const float* A, int K, const float* Wt, float* C, const float* bias,
                   long M, int N, int act) {
        gemm_full(stream, A, K, 1, 0, Wt, K, 1, 0, C, N, 0, bias, (int)M, N, K, 1, 1.0f, act, 0, 0);
    };

    // =================== level encoders ===================
    {   // AA: Linear(64->128)+ReLU -> BN -> Linear(128->256) -> LN
        off = scratch0;
        float* h1  = alloc((size_t)B * LAA * 128);
        float* pre = alloc((size_t)B * LAA * H);
        lin(aaF, 64, P[0], h1, P[1], (long)B * LAA, 128, ACT_RELU);
        bn_kernel<<<128, 256, 0, stream>>>(h1, P[2], P[3], B * LAA, 128);
        lin(h1, 128, P[4], pre, P[5], (long)B * LAA, 256, ACT_NONE);
        ln_kernel<<<B * LAA, 256, 0, stream>>>(pre, aaN, P[6], P[7], B * LAA, B * LAA, 0);
    }
    {   // SS: conv1d(128->256,k3)+ReLU -> BN -> conv1d(256->256,k3) -> LN over length
        off = scratch0;
        float* x1 = alloc((size_t)B * LSS * H);   // stored [B, L, C]
        for (int d = 0; d < 3; ++d)
            gemm_full(stream, ssF, 128, 1, (long)LSS * 128,
                      P[8] + d, 128 * 3, 3, 0,
                      x1, 256, (long)LSS * 256,
                      (d == 0) ? P[9] : nullptr,
                      LSS, 256, 128, B, 1.0f, (d == 2) ? ACT_RELU : ACT_NONE, (d > 0) ? 1 : 0, d - 1);
        bn_kernel<<<256, 256, 0, stream>>>(x1, P[10], P[11], B * LSS, 256);
        for (int d = 0; d < 3; ++d)
            gemm_full(stream, x1, 256, 1, (long)LSS * 256,
                      P[12] + d, 256 * 3, 3, 0,
                      ssN, 256, (long)LSS * 256,
                      (d == 0) ? P[13] : nullptr,
                      LSS, 256, 256, B, 1.0f, ACT_NONE, (d > 0) ? 1 : 0, d - 1);
        ln_axis_kernel<<<B * 256, 256, 0, stream>>>(ssN, P[14], P[15], LSS, 256);
    }
    {   // DOM
        off = scratch0;
        float* h1  = alloc((size_t)B * LDOM * 256);
        float* pre = alloc((size_t)B * LDOM * 256);
        lin(domF, 256, P[16], h1, P[17], (long)B * LDOM, 256, ACT_RELU);
        lin(h1, 256, P[18], pre, P[19], (long)B * LDOM, 256, ACT_NONE);
        ln_kernel<<<B * LDOM, 256, 0, stream>>>(pre, domN, P[20], P[21], B * LDOM, B * LDOM, 0);
    }
    {   // PROT
        off = scratch0;
        float* h1  = alloc((size_t)B * 512);
        float* pre = alloc((size_t)B * 256);
        lin(prF, 512, P[22], h1, P[23], (long)B, 512, ACT_RELU);
        lin(h1, 512, P[24], pre, P[25], (long)B, 256, ACT_NONE);
        ln_kernel<<<B, 256, 0, stream>>>(pre, protN, P[26], P[27], B, B, 0);
    }

    // =================== message passing ===================
    auto pass = [&](const float* src, const float* tgt, const int* si, const int* ti,
                    int E, int Ls, int Lt, int mp, int gp, float* outN) {
        off = scratch0;
        long RE = (long)B * E, RT = (long)B * Lt;
        float* comb = alloc(RE * 512);
        float* hid  = alloc(RE * 256);
        float* msg  = alloc(RE * 256);
        float* msgs = alloc(RT * 256);
        float* gi   = alloc(RT * 768);
        float* gh   = alloc(RT * 768);
        long tot = RE * 512;
        gather_concat_kernel<<<gb(tot), 256, 0, stream>>>(src, tgt, si, ti, comb, E, Ls, Lt, tot);
        lin(comb, 512, P[mp + 0], hid, P[mp + 1], RE, 256, ACT_RELU);
        lin(hid, 256, P[mp + 2], msg, P[mp + 3], RE, 256, ACT_NONE);
        zero_kernel<<<gb(RT * 256), 256, 0, stream>>>(msgs, RT * 256);
        scatter_add_kernel<<<gb(RE * 256), 256, 0, stream>>>(msg, ti, msgs, E, Lt, RE * 256);
        lin(msgs, 256, P[gp + 0], gi, P[gp + 2], RT, 768, ACT_NONE);
        lin(tgt,  256, P[gp + 1], gh, P[gp + 3], RT, 768, ACT_NONE);
        gru_kernel<<<gb(RT * 256), 256, 0, stream>>>(gi, gh, tgt, outN, RT);
    };

    pass(aaN,    ssN,    edge[0],  edge[1],  2048, LAA,  LSS,  28, 52, buSS);   // bu aa->ss
    pass(buSS,   domN,   edge[2],  edge[3],  256,  LSS,  LDOM, 32, 56, buDOM);  // bu ss->dom
    pass(buDOM,  protN,  edge[4],  edge[5],  32,   LDOM, 1,    36, 60, buPROT); // bu dom->prot
    pass(buPROT, buDOM,  edge[6],  edge[7],  32,   1,    LDOM, 40, 64, tdDOM);  // td prot->dom
    pass(tdDOM,  buSS,   edge[8],  edge[9],  256,  LDOM, LSS,  44, 68, tdSS);   // td dom->ss
    pass(tdSS,   aaN,    edge[10], edge[11], 2048, LSS,  LAA,  48, 72, tdAA);   // td ss->aa

    // =================== bidirectional integrators ===================
    const float ascale = 1.0f / sqrtf(32.0f);
    auto integ = [&](const float* bu, const float* td, int L, int rowOff) {
        off = scratch0;
        long R = (long)B * L;
        float* comb   = alloc(R * 512);           // reused as fc
        float* gbu    = alloc(R * 256);
        float* gtd    = alloc(R * 256);
        float* q      = alloc(R * 256);           // reused as ao
        float* k      = alloc(R * 256);           // reused as hid
        float* v      = alloc(R * 256);           // reused as pre
        float* att    = alloc(R * 256);
        float* scores = alloc((long)L * L);

        concat2_kernel<<<gb(R * 512), 256, 0, stream>>>(bu, td, comb, R);
        lin(comb, 512, P[80], gbu, P[81], R, 256, ACT_NONE);
        gate_kernel<<<gb(R * 256), 256, 0, stream>>>(bu, gbu, R * 256);
        lin(comb, 512, P[82], gtd, P[83], R, 256, ACT_NONE);
        gate_kernel<<<gb(R * 256), 256, 0, stream>>>(td, gtd, R * 256);

        // MHA projections (slices of in_proj)
        lin(gbu, 256, P[76],            q, P[77],       R, 256, ACT_NONE);
        lin(gtd, 256, P[76] + 256 * 256, k, P[77] + 256, R, 256, ACT_NONE);
        lin(gtd, 256, P[76] + 512 * 256, v, P[77] + 512, R, 256, ACT_NONE);

        for (int b = 0; b < B; ++b) {
            for (int h = 0; h < 8; ++h) {
                const float* qb = q + (long)b * L * 256 + h * 32;
                const float* kb = k + (long)b * L * 256 + h * 32;
                const float* vb = v + (long)b * L * 256 + h * 32;
                float* ab = att + (long)b * L * 256 + h * 32;
                // scores = (Q @ K^T) / sqrt(dh)
                gemm_full(stream, qb, 256, 1, 0, kb, 256, 1, 0,
                          scores, L, 0, nullptr, L, L, 32, 1, ascale, ACT_NONE, 0, 0);
                softmax_kernel<<<L, 256, 0, stream>>>(scores, L);
                // out = attn @ V  (B addressed with swapped strides)
                gemm_full(stream, scores, L, 1, 0, vb, 1, 256, 0,
                          ab, 256, 0, nullptr, L, 32, L, 1, 1.0f, ACT_NONE, 0, 0);
            }
        }
        float* ao  = q;
        float* fc  = comb;
        float* hid = k;
        float* pre = v;
        lin(att, 256, P[78], ao, P[79], R, 256, ACT_NONE);
        fccat_kernel<<<gb(R * 512), 256, 0, stream>>>(ao, gbu, gtd, fc, R);
        lin(fc, 512, P[84], hid, P[85], R, 256, ACT_RELU);
        lin(hid, 256, P[86], pre, P[87], R, 256, ACT_NONE);
        ln_kernel<<<(int)R, 256, 0, stream>>>(pre, out + (long)rowOff * 256,
                                              P[88], P[89], (int)R, L, ROWT * 256);
    };

    integ(aaN,    tdAA,  LAA,  0);
    integ(buSS,   tdSS,  LSS,  LAA);
    integ(buDOM,  tdDOM, LDOM, LAA + LSS);
    integ(buPROT, buPROT, 1,   LAA + LSS + LDOM);
}
